// BiDAF_10599979286604
// MI455X (gfx1250) — compile-verified, matrix operand order checked by reading, not run on hardware
//
#include <hip/hip_runtime.h>
#include <math.h>

// ---------------------------------------------------------------------------
// Types for CDNA5 WMMA (wave32): D(16x16,f32) = A(16x32,bf16) x B(32x16,bf16) + C
// ---------------------------------------------------------------------------
typedef __attribute__((ext_vector_type(16))) __bf16 v16bf;
typedef __attribute__((ext_vector_type(8)))  __bf16 v8bf;
typedef __attribute__((ext_vector_type(8)))  float  v8f;

// Fragment load matching the documented 16-bit A/B layout:
//  lanes 0-15: elements 0..7 = K(base+0..7), 8..15 = K(base+16..23), base=0
//  lanes16-31: same with base=8. Caller pre-offsets pointer by (lane&16)?8:0.
__device__ __forceinline__ v16bf ld_frag(const __bf16* p) {
    v8bf lo = *reinterpret_cast<const v8bf*>(p);
    v8bf hi = *reinterpret_cast<const v8bf*>(p + 16);
    v16bf r;
#pragma unroll
    for (int i = 0; i < 8; ++i) { r[i] = lo[i]; r[8 + i] = hi[i]; }
    return r;
}

__device__ __forceinline__ v8f wmma_bf16(v16bf a, v16bf b, v8f c) {
    return __builtin_amdgcn_wmma_f32_16x16x32_bf16(
        /*neg_a=*/false, a, /*neg_b=*/false, b,
        /*c_mod=*/(short)0, c, /*reuse_a=*/false, /*reuse_b=*/false);
}

__device__ __forceinline__ float sigmf(float x) { return 1.0f / (1.0f + __expf(-x)); }

// ---------------------------------------------------------------------------
// Generic batched GEMM:  C[b](M x N) = A[b](M x Kp, bf16) * B[b](N x Kp, bf16)^T
//                         (+bias[n]) (+epilogue)
// Kp multiple of 32; M, N multiples of 16. Block = 256 thr = 8 waves (2Mx4N),
// each wave computes 64x32 (4x2 tiles of 16x16). EPI: 0 none, 1 relu, 2 sigmoid.
// L2-resident operands; no LDS staging (MI455X: 192MB L2, 23.3TB/s HBM).
// ---------------------------------------------------------------------------
template <int EPI>
__global__ __launch_bounds__(256) void gemm_wmma(
    const __bf16* __restrict__ Abase, long long aBatch,
    const __bf16* __restrict__ Bbase, long long bBatch,
    const float* __restrict__ bias,
    float* __restrict__ Cbase, long long cBatch,
    int M, int N, int Kp, int ldc)
{
    const __bf16* Ap = Abase + (long long)blockIdx.z * aBatch;
    const __bf16* Bp = Bbase + (long long)blockIdx.z * bBatch;
    float*        Cp = Cbase + (long long)blockIdx.z * cBatch;

    int wave = threadIdx.x >> 5;
    int lane = threadIdx.x & 31;
    int m0 = blockIdx.y * 128 + (wave >> 2) * 64;
    int n0 = blockIdx.x * 128 + (wave & 3) * 32;
    if (m0 >= M || n0 >= N) return;

    bool mOk[4], nOk[2];
#pragma unroll
    for (int mt = 0; mt < 4; ++mt) mOk[mt] = (m0 + mt * 16 + 16) <= M;
#pragma unroll
    for (int nt = 0; nt < 2; ++nt) nOk[nt] = (n0 + nt * 16 + 16) <= N;

    int lrow = lane & 15;
    int kb   = (lane & 16) ? 8 : 0;
    const __bf16* arow[4];
    const __bf16* brow[2];
#pragma unroll
    for (int mt = 0; mt < 4; ++mt) arow[mt] = Ap + (long long)(m0 + mt * 16 + lrow) * Kp + kb;
#pragma unroll
    for (int nt = 0; nt < 2; ++nt) brow[nt] = Bp + (long long)(n0 + nt * 16 + lrow) * Kp + kb;

    v8f acc[4][2] = {};
    for (int k = 0; k < Kp; k += 32) {
        if (k + 128 < Kp) {
            __builtin_prefetch(arow[0] + k + 128);
            __builtin_prefetch(brow[0] + k + 128);
        }
        v16bf af[4], bfr[2];
#pragma unroll
        for (int mt = 0; mt < 4; ++mt) if (mOk[mt]) af[mt] = ld_frag(arow[mt] + k);
#pragma unroll
        for (int nt = 0; nt < 2; ++nt) if (nOk[nt]) bfr[nt] = ld_frag(brow[nt] + k);
#pragma unroll
        for (int mt = 0; mt < 4; ++mt)
#pragma unroll
            for (int nt = 0; nt < 2; ++nt)
                if (mOk[mt] && nOk[nt]) acc[mt][nt] = wmma_bf16(af[mt], bfr[nt], acc[mt][nt]);
    }

    // C/D layout: VGPR e -> row = e + (lane>=16 ? 8 : 0), col = lane&15
    int cc = lane & 15;
    int rb = (lane & 16) ? 8 : 0;
#pragma unroll
    for (int mt = 0; mt < 4; ++mt) {
#pragma unroll
        for (int nt = 0; nt < 2; ++nt) {
            if (!(mOk[mt] && nOk[nt])) continue;
            int col = n0 + nt * 16 + cc;
            float bv = bias ? bias[col] : 0.0f;
#pragma unroll
            for (int e = 0; e < 8; ++e) {
                int row = m0 + mt * 16 + rb + e;
                float v = acc[mt][nt][e] + bv;
                if (EPI == 1) v = fmaxf(v, 0.0f);
                else if (EPI == 2) v = sigmf(v);
                Cp[(long long)row * ldc + col] = v;
            }
        }
    }
}

// ---------------------------------------------------------------------------
// Persistent bidirectional GRU recurrence. grid.x = 2 (dir), 256 thr = 8 waves.
// Per step: gh = h(32x400) * Whh(1200x416 bf16)^T via WMMA, gates in registers.
// Hidden state double-buffered in LDS (bf16). xp (with bih) precomputed.
// Output written into out[b][t][dir*400 + j], ldo = 800.
// ---------------------------------------------------------------------------
__global__ __launch_bounds__(256) void gru_seq(
    const float* __restrict__ xp_f, const float* __restrict__ xp_b,
    const __bf16* __restrict__ Whh_f, const __bf16* __restrict__ Whh_b,
    const float* __restrict__ bhh_f, const float* __restrict__ bhh_b,
    float* __restrict__ out, int T, int ldo)
{
    const int KP = 416;   // padded K (400 -> 416, zero filled)
    const int HP = 424;   // LDS row stride in halves (16B aligned)
    __shared__ __bf16 hbuf[2][32][HP];

    int dir = blockIdx.x;
    const float*  xp  = dir ? xp_b  : xp_f;
    const __bf16* Whh = dir ? Whh_b : Whh_f;
    const float*  bhh = dir ? bhh_b : bhh_f;

    for (int i = threadIdx.x; i < 2 * 32 * HP; i += blockDim.x)
        (&hbuf[0][0][0])[i] = (__bf16)0.0f;
    __syncthreads();

    int wave = threadIdx.x >> 5;
    int lane = threadIdx.x & 31;
    int lrow = lane & 15;
    int kb   = (lane & 16) ? 8 : 0;
    int cc   = lane & 15;
    int rb   = (lane & 16) ? 8 : 0;

    int cur = 0;
    for (int s = 0; s < T; ++s) {
        int t = dir ? (T - 1 - s) : s;
        int nxt = cur ^ 1;
        for (int strip = wave; strip < 25; strip += 8) {   // 25 strips of 16 cols
            int j0 = strip * 16;
            const __bf16* a0p = &hbuf[cur][lrow][kb];
            const __bf16* a1p = &hbuf[cur][16 + lrow][kb];
            const __bf16* brp = Whh + (long long)(j0 + lrow) * KP + kb;
            const __bf16* bzp = Whh + (long long)(j0 + 400 + lrow) * KP + kb;
            const __bf16* bnp = Whh + (long long)(j0 + 800 + lrow) * KP + kb;
            v8f acc[2][3] = {};
            for (int k = 0; k < KP; k += 32) {
                v16bf a0 = ld_frag(a0p + k);
                v16bf a1 = ld_frag(a1p + k);
                v16bf br = ld_frag(brp + k);
                v16bf bz = ld_frag(bzp + k);
                v16bf bn = ld_frag(bnp + k);
                acc[0][0] = wmma_bf16(a0, br, acc[0][0]);
                acc[1][0] = wmma_bf16(a1, br, acc[1][0]);
                acc[0][1] = wmma_bf16(a0, bz, acc[0][1]);
                acc[1][1] = wmma_bf16(a1, bz, acc[1][1]);
                acc[0][2] = wmma_bf16(a0, bn, acc[0][2]);
                acc[1][2] = wmma_bf16(a1, bn, acc[1][2]);
            }
            int j = j0 + cc;
            float brb = bhh[j], bzb = bhh[j + 400], bnb = bhh[j + 800];
#pragma unroll
            for (int mt = 0; mt < 2; ++mt) {
#pragma unroll
                for (int e = 0; e < 8; ++e) {
                    int b = mt * 16 + rb + e;
                    const float* xrow = xp + ((long long)b * T + t) * 1200;
                    float r = sigmf(xrow[j]       + acc[mt][0][e] + brb);
                    float z = sigmf(xrow[j + 400] + acc[mt][1][e] + bzb);
                    float n = tanhf(xrow[j + 800] + r * (acc[mt][2][e] + bnb));
                    float hp = (float)hbuf[cur][b][j];
                    float hv = (1.0f - z) * n + z * hp;
                    hbuf[nxt][b][j] = (__bf16)hv;
                    out[((long long)b * T + t) * ldo + dir * 400 + j] = hv;
                }
            }
        }
        __syncthreads();
        cur ^= 1;
    }
}

// ---------------------------------------------------------------------------
// Conversion / elementwise kernels
// ---------------------------------------------------------------------------
__global__ __launch_bounds__(256) void cvt_pad_bf16(
    const float* __restrict__ src, __bf16* __restrict__ dst, long long rows, int K, int Kp)
{
    long long i = (long long)blockIdx.x * blockDim.x + threadIdx.x;
    if (i >= rows * (long long)Kp) return;
    long long r = i / Kp;
    int k = (int)(i % Kp);
    dst[i] = (k < K) ? (__bf16)src[r * K + k] : (__bf16)0.0f;
}

// src is (K,N) row-major f32 -> dst (N,Kp) row-major bf16 (for x @ W layouts)
__global__ __launch_bounds__(256) void cvt_kn_to_nk_bf16(
    const float* __restrict__ src, __bf16* __restrict__ dst, int K, int N, int Kp)
{
    long long i = (long long)blockIdx.x * blockDim.x + threadIdx.x;
    if (i >= (long long)N * Kp) return;
    int n = (int)(i / Kp);
    int k = (int)(i % Kp);
    dst[i] = (k < K) ? (__bf16)src[(long long)k * N + n] : (__bf16)0.0f;
}

__global__ __launch_bounds__(256) void scale_cvt_bf16(
    const float* __restrict__ x, const float* __restrict__ w,
    __bf16* __restrict__ dst, long long rows, int K)
{
    long long i = (long long)blockIdx.x * blockDim.x + threadIdx.x;
    if (i >= rows * (long long)K) return;
    dst[i] = (__bf16)(x[i] * w[(int)(i % K)]);
}

// src (B,R,C) f32 -> dst (B,C,R) bf16
__global__ __launch_bounds__(256) void transpose_bf16_batched(
    const float* __restrict__ src, __bf16* __restrict__ dst, int B, int R, int C)
{
    long long i = (long long)blockIdx.x * blockDim.x + threadIdx.x;
    long long tot = (long long)B * R * C;
    if (i >= tot) return;
    int b = (int)(i / ((long long)R * C));
    long long rem = i % ((long long)R * C);
    int r = (int)(rem / C);
    int c = (int)(rem % C);
    dst[((long long)b * C + c) * R + r] = (__bf16)src[i];
}

__global__ __launch_bounds__(256) void word_gather(
    const int* __restrict__ words, const float* __restrict__ emb,
    float* __restrict__ x, long long nPos)
{
    long long i = (long long)blockIdx.x * blockDim.x + threadIdx.x;
    if (i >= nPos * 300) return;
    long long pos = i / 300;
    int d = (int)(i % 300);
    x[pos * 400 + 100 + d] = emb[(long long)words[pos] * 300 + d];
}

// Char CNN: one block per word. chars (16,), conv_w (100,100,5), out max_p relu.
__global__ __launch_bounds__(128) void char_conv(
    const int* __restrict__ chars, const float* __restrict__ emb,
    const float* __restrict__ w, const float* __restrict__ bias,
    float* __restrict__ x)
{
    __shared__ float e[16][100];
    long long word = blockIdx.x;
    const int* ch = chars + word * 16;
    for (int i = threadIdx.x; i < 1600; i += 128) {
        int p = i / 100, c = i % 100;
        e[p][c] = emb[(long long)ch[p] * 100 + c];
    }
    __syncthreads();
    int o = threadIdx.x;
    if (o < 100) {
        float acc[12];
#pragma unroll
        for (int p = 0; p < 12; ++p) acc[p] = 0.0f;
        const float* wo = w + o * 500;
        for (int i = 0; i < 100; ++i) {
#pragma unroll
            for (int k = 0; k < 5; ++k) {
                float wv = wo[i * 5 + k];
#pragma unroll
                for (int p = 0; p < 12; ++p) acc[p] += wv * e[p + k][i];
            }
        }
        float b = bias[o];
        float mx = 0.0f;  // relu: max of nonneg values, 0 floor
#pragma unroll
        for (int p = 0; p < 12; ++p) mx = fmaxf(mx, acc[p] + b);
        x[word * 400 + o] = mx;
    }
}

__global__ __launch_bounds__(256) void hw_combine(
    const float* __restrict__ x, const float* __restrict__ t,
    const float* __restrict__ g, float* __restrict__ y, long long n)
{
    long long i = (long long)blockIdx.x * blockDim.x + threadIdx.x;
    if (i >= n) return;
    float gv = g[i];
    y[i] = gv * t[i] + (1.0f - gv) * x[i];
}

__global__ __launch_bounds__(256) void rowdot(
    const float* __restrict__ X, const float* __restrict__ w,
    float* __restrict__ out, long long rows, int K)
{
    long long r = (long long)blockIdx.x * blockDim.x + threadIdx.x;
    if (r >= rows) return;
    const float* xr = X + r * K;
    float s = 0.0f;
    for (int k = 0; k < K; ++k) s += xr[k] * w[k];
    out[r] = s;
}

__global__ __launch_bounds__(256) void sim_add(
    float* __restrict__ sim, const float* __restrict__ cwh,
    const float* __restrict__ qwu, int B, int T, int J)
{
    long long i = (long long)blockIdx.x * blockDim.x + threadIdx.x;
    if (i >= (long long)B * T * J) return;
    int j = (int)(i % J);
    long long bt = i / J;
    int b = (int)(bt / T);
    sim[i] += cwh[bt] + qwu[(long long)b * J + j];
}

__global__ __launch_bounds__(256) void rowmax_J(
    const float* __restrict__ sim, float* __restrict__ smax, long long BT, int J)
{
    long long bt = (long long)blockIdx.x * blockDim.x + threadIdx.x;
    if (bt >= BT) return;
    const float* row = sim + bt * J;
    float m = row[0];
    for (int j = 1; j < J; ++j) m = fmaxf(m, row[j]);
    smax[bt] = m;
}

// softmax over T (axis=1) for each (b,j); in-place on sim
__global__ __launch_bounds__(256) void softmax_T_bj(
    float* __restrict__ sim, int B, int T, int J)
{
    int idx = blockIdx.x * blockDim.x + threadIdx.x;
    if (idx >= B * J) return;
    int b = idx / J, j = idx % J;
    float* base = sim + (long long)b * T * J + j;
    float mx = -1e30f;
    for (int t = 0; t < T; ++t) mx = fmaxf(mx, base[(long long)t * J]);
    float s = 0.0f;
    for (int t = 0; t < T; ++t) s += __expf(base[(long long)t * J] - mx);
    float inv = 1.0f / s;
    for (int t = 0; t < T; ++t) {
        float v = __expf(base[(long long)t * J] - mx) * inv;
        base[(long long)t * J] = v;
    }
}

// softmax over T per batch row (one lane per b)
__global__ __launch_bounds__(32) void softmax_T_b(
    const float* __restrict__ s, float* __restrict__ p, int B, int T)
{
    int b = threadIdx.x;
    if (b >= B) return;
    const float* row = s + (long long)b * T;
    float* pr = p + (long long)b * T;
    float mx = -1e30f;
    for (int t = 0; t < T; ++t) mx = fmaxf(mx, row[t]);
    float sum = 0.0f;
    for (int t = 0; t < T; ++t) sum += __expf(row[t] - mx);
    float inv = 1.0f / sum;
    for (int t = 0; t < T; ++t) pr[t] = __expf(row[t] - mx) * inv;
}

__global__ __launch_bounds__(256) void weighted_sum_T(
    const float* __restrict__ attb, const float* __restrict__ c,
    float* __restrict__ ht, int B, int T, int Dd)
{
    long long i = (long long)blockIdx.x * blockDim.x + threadIdx.x;
    if (i >= (long long)B * Dd) return;
    int b = (int)(i / Dd);
    int d = (int)(i % Dd);
    float s = 0.0f;
    for (int t = 0; t < T; ++t)
        s += attb[(long long)b * T + t] * c[((long long)b * T + t) * Dd + d];
    ht[i] = s;
}

__global__ __launch_bounds__(256) void build_g(
    const float* __restrict__ c, const float* __restrict__ ut,
    const float* __restrict__ ht, float* __restrict__ g, int B, int T, int Dd)
{
    long long i = (long long)blockIdx.x * blockDim.x + threadIdx.x;
    if (i >= (long long)B * T * Dd) return;
    long long bt = i / Dd;
    int d = (int)(i % Dd);
    int b = (int)(bt / T);
    float cv = c[i], uv = ut[i];
    long long base = bt * (4LL * Dd);
    g[base + d]          = cv;
    g[base + Dd + d]     = uv;
    g[base + 2 * Dd + d] = cv * uv;
    g[base + 3 * Dd + d] = cv * ht[(long long)b * Dd + d];
}

__global__ __launch_bounds__(128) void score_dot(
    const float* __restrict__ g, const float* __restrict__ m,
    const float* __restrict__ w, float* __restrict__ s, long long rows)
{
    long long r = (long long)blockIdx.x * blockDim.x + threadIdx.x;
    if (r >= rows) return;
    const float* gr = g + r * 3200;
    const float* mr = m + r * 800;
    float acc = 0.0f;
    for (int k = 0; k < 3200; ++k) acc += gr[k] * w[k];
    for (int k = 0; k < 800; ++k)  acc += mr[k] * w[3200 + k];
    s[r] = acc;
}

// ---------------------------------------------------------------------------
// Host orchestration
// ---------------------------------------------------------------------------
extern "C" void kernel_launch(void* const* d_in, const int* in_sizes, int n_in,
                              void* d_out, int out_size, void* d_ws, size_t ws_size,
                              hipStream_t stream)
{
    (void)in_sizes; (void)n_in; (void)out_size; (void)ws_size;

    const int B = 32, T = 512, J = 64;
    const int KD = 416;  // 400 padded to multiple of 32
    const long long BT = (long long)B * T;  // 16384
    const long long BJ = (long long)B * J;  // 2048

    // ---- inputs (setup_inputs dict order; GRU dicts flatten Wih,Whh,bih,bhh)
    const int*   word_c   = (const int*)d_in[0];
    const int*   char_c   = (const int*)d_in[1];
    const int*   word_q   = (const int*)d_in[2];
    const int*   char_q   = (const int*)d_in[3];
    const float* char_emb = (const float*)d_in[4];
    const float* conv_w   = (const float*)d_in[5];
    const float* conv_b   = (const float*)d_in[6];
    const float* word_emb = (const float*)d_in[7];
    const float* hw_t_w   = (const float*)d_in[8];
    const float* hw_t_b   = (const float*)d_in[9];
    const float* hw_g_w   = (const float*)d_in[10];
    const float* hw_g_b   = (const float*)d_in[11];
    const float* ctxF_Wih = (const float*)d_in[12];
    const float* ctxF_Whh = (const float*)d_in[13];
    const float* ctxF_bih = (const float*)d_in[14];
    const float* ctxF_bhh = (const float*)d_in[15];
    const float* ctxB_Wih = (const float*)d_in[16];
    const float* ctxB_Whh = (const float*)d_in[17];
    const float* ctxB_bih = (const float*)d_in[18];
    const float* ctxB_bhh = (const float*)d_in[19];
    const float* w_s      = (const float*)d_in[20];
    const float* m1F_Wih  = (const float*)d_in[21];
    const float* m1F_Whh  = (const float*)d_in[22];
    const float* m1F_bih  = (const float*)d_in[23];
    const float* m1F_bhh  = (const float*)d_in[24];
    const float* m1B_Wih  = (const float*)d_in[25];
    const float* m1B_Whh  = (const float*)d_in[26];
    const float* m1B_bih  = (const float*)d_in[27];
    const float* m1B_bhh  = (const float*)d_in[28];
    const float* m2F_Wih  = (const float*)d_in[29];
    const float* m2F_Whh  = (const float*)d_in[30];
    const float* m2F_bih  = (const float*)d_in[31];
    const float* m2F_bhh  = (const float*)d_in[32];
    const float* m2B_Wih  = (const float*)d_in[33];
    const float* m2B_Whh  = (const float*)d_in[34];
    const float* m2B_bih  = (const float*)d_in[35];
    const float* m2B_bhh  = (const float*)d_in[36];
    const float* w_p_start= (const float*)d_in[37];
    const float* oF_Wih   = (const float*)d_in[38];
    const float* oF_Whh   = (const float*)d_in[39];
    const float* oF_bih   = (const float*)d_in[40];
    const float* oF_bhh   = (const float*)d_in[41];
    const float* oB_Wih   = (const float*)d_in[42];
    const float* oB_Whh   = (const float*)d_in[43];
    const float* oB_bih   = (const float*)d_in[44];
    const float* oB_bhh   = (const float*)d_in[45];
    const float* w_p_end  = (const float*)d_in[46];

    // ---- workspace bump allocator
    char* wsBase = (char*)d_ws;
    size_t off = 0;
    auto alloc = [&](size_t bytes) -> void* {
        void* p = wsBase + off;
        off = (off + bytes + 255) & ~(size_t)255;
        return p;
    };

    float*  x_c    = (float*)alloc(BT * 400 * 4);
    float*  x_q    = (float*)alloc(BJ * 400 * 4);
    float*  t_c    = (float*)alloc(BT * 400 * 4);
    float*  gate_c = (float*)alloc(BT * 400 * 4);
    float*  xh_c   = (float*)alloc(BT * 400 * 4);
    float*  t_q    = (float*)alloc(BJ * 400 * 4);
    float*  gate_q = (float*)alloc(BJ * 400 * 4);
    float*  xh_q   = (float*)alloc(BJ * 400 * 4);
    __bf16* xc_bf  = (__bf16*)alloc(BT * KD * 2);
    __bf16* xq_bf  = (__bf16*)alloc(BJ * KD * 2);
    float*  xp_f   = (float*)alloc(BT * 1200 * 4);
    float*  xp_b   = (float*)alloc(BT * 1200 * 4);
    float*  c_ctx  = (float*)alloc(BT * 800 * 4);
    float*  q_ctx  = (float*)alloc(BJ * 800 * 4);
    __bf16* cw_bf  = (__bf16*)alloc(BT * 800 * 2);
    __bf16* q_bf   = (__bf16*)alloc(BJ * 800 * 2);
    float*  cwh    = (float*)alloc(BT * 4);
    float*  qwu    = (float*)alloc(BJ * 4);
    float*  sim    = (float*)alloc((long long)B * T * J * 4);
    __bf16* atta_bf= (__bf16*)alloc((long long)B * T * J * 2);
    __bf16* qT_bf  = (__bf16*)alloc((long long)B * 800 * J * 2);
    float*  u_t    = (float*)alloc(BT * 800 * 4);
    float*  smax   = (float*)alloc(BT * 4);
    float*  attb   = (float*)alloc(BT * 4);
    float*  h_t    = (float*)alloc((long long)B * 800 * 4);
    float*  gbuf   = (float*)alloc(BT * 3200 * 4);
    __bf16* g_bf   = (__bf16*)alloc(BT * 3200 * 2);
    float*  m1     = (float*)alloc(BT * 800 * 4);
    __bf16* m1_bf  = (__bf16*)alloc(BT * 800 * 2);
    float*  mbuf   = (float*)alloc(BT * 800 * 4);
    __bf16* m_bf   = (__bf16*)alloc(BT * 800 * 2);
    float*  m2     = (float*)alloc(BT * 800 * 4);
    float*  s1     = (float*)alloc(BT * 4);
    float*  s2     = (float*)alloc(BT * 4);

    __bf16* hwT_bf  = (__bf16*)alloc(400 * KD * 2);
    __bf16* hwG_bf  = (__bf16*)alloc(400 * KD * 2);
    __bf16* WihCtxF = (__bf16*)alloc(1200 * KD * 2);
    __bf16* WihCtxB = (__bf16*)alloc(1200 * KD * 2);
    __bf16* WhhCtxF = (__bf16*)alloc(1200 * KD * 2);
    __bf16* WhhCtxB = (__bf16*)alloc(1200 * KD * 2);
    __bf16* WihM1F  = (__bf16*)alloc(1200LL * 3200 * 2);
    __bf16* WihM1B  = (__bf16*)alloc(1200LL * 3200 * 2);
    __bf16* WhhM1F  = (__bf16*)alloc(1200 * KD * 2);
    __bf16* WhhM1B  = (__bf16*)alloc(1200 * KD * 2);
    __bf16* WihM2F  = (__bf16*)alloc(1200 * 800 * 2);
    __bf16* WihM2B  = (__bf16*)alloc(1200 * 800 * 2);
    __bf16* WhhM2F  = (__bf16*)alloc(1200 * KD * 2);
    __bf16* WhhM2B  = (__bf16*)alloc(1200 * KD * 2);
    __bf16* WihOF   = (__bf16*)alloc(1200 * 800 * 2);
    __bf16* WihOB   = (__bf16*)alloc(1200 * 800 * 2);
    __bf16* WhhOF   = (__bf16*)alloc(1200 * KD * 2);
    __bf16* WhhOB   = (__bf16*)alloc(1200 * KD * 2);

    const int THR = 256;
    auto nb = [](long long n, int b) { return (unsigned)((n + b - 1) / b); };

    auto gemm = [&](int epi, const __bf16* A_, long long aB, const __bf16* B_, long long bB,
                    const float* bias, float* C_, long long cB,
                    int M, int N, int Kp, int ldc, int batch) {
        dim3 grid((N + 127) / 128, (M + 127) / 128, batch);
        if (epi == 0)      gemm_wmma<0><<<grid, 256, 0, stream>>>(A_, aB, B_, bB, bias, C_, cB, M, N, Kp, ldc);
        else if (epi == 1) gemm_wmma<1><<<grid, 256, 0, stream>>>(A_, aB, B_, bB, bias, C_, cB, M, N, Kp, ldc);
        else               gemm_wmma<2><<<grid, 256, 0, stream>>>(A_, aB, B_, bB, bias, C_, cB, M, N, Kp, ldc);
    };
    auto gru = [&](const float* xpf, const float* xpb, const __bf16* whf, const __bf16* whb,
                   const float* bhf, const float* bhb, float* outp, int Tlen) {
        gru_seq<<<dim3(2), dim3(256), 0, stream>>>(xpf, xpb, whf, whb, bhf, bhb, outp, Tlen, 800);
    };

    // ---- weight conversions to bf16 (K padded to mult of 32, zero filled)
    cvt_kn_to_nk_bf16<<<nb(400LL * KD, THR), THR, 0, stream>>>(hw_t_w, hwT_bf, 400, 400, KD);
    cvt_kn_to_nk_bf16<<<nb(400LL * KD, THR), THR, 0, stream>>>(hw_g_w, hwG_bf, 400, 400, KD);
    struct NKW { const float* s; __bf16* d; int N; int K; int Kp; };
    NKW nkw[] = {
        {ctxF_Wih, WihCtxF, 1200, 400, KD}, {ctxB_Wih, WihCtxB, 1200, 400, KD},
        {ctxF_Whh, WhhCtxF, 1200, 400, KD}, {ctxB_Whh, WhhCtxB, 1200, 400, KD},
        {m1F_Wih,  WihM1F,  1200, 3200, 3200}, {m1B_Wih, WihM1B, 1200, 3200, 3200},
        {m1F_Whh,  WhhM1F,  1200, 400, KD}, {m1B_Whh, WhhM1B, 1200, 400, KD},
        {m2F_Wih,  WihM2F,  1200, 800, 800}, {m2B_Wih, WihM2B, 1200, 800, 800},
        {m2F_Whh,  WhhM2F,  1200, 400, KD}, {m2B_Whh, WhhM2B, 1200, 400, KD},
        {oF_Wih,   WihOF,   1200, 800, 800}, {oB_Wih,  WihOB,  1200, 800, 800},
        {oF_Whh,   WhhOF,   1200, 400, KD}, {oB_Whh,  WhhOB,  1200, 400, KD},
    };
    for (auto& w : nkw)
        cvt_pad_bf16<<<nb((long long)w.N * w.Kp, THR), THR, 0, stream>>>(w.s, w.d, w.N, w.K, w.Kp);

    // ---- embeddings: char CNN + word gather -> x (.,400)
    char_conv<<<(unsigned)BT, 128, 0, stream>>>(char_c, char_emb, conv_w, conv_b, x_c);
    char_conv<<<(unsigned)BJ, 128, 0, stream>>>(char_q, char_emb, conv_w, conv_b, x_q);
    word_gather<<<nb(BT * 300, THR), THR, 0, stream>>>(word_c, word_emb, x_c, BT);
    word_gather<<<nb(BJ * 300, THR), THR, 0, stream>>>(word_q, word_emb, x_q, BJ);

    // ---- highway
    cvt_pad_bf16<<<nb(BT * KD, THR), THR, 0, stream>>>(x_c, xc_bf, BT, 400, KD);
    cvt_pad_bf16<<<nb(BJ * KD, THR), THR, 0, stream>>>(x_q, xq_bf, BJ, 400, KD);
    gemm(1, xc_bf, 0, hwT_bf, 0, hw_t_b, t_c,    0, (int)BT, 400, KD, 400, 1);
    gemm(2, xc_bf, 0, hwG_bf, 0, hw_g_b, gate_c, 0, (int)BT, 400, KD, 400, 1);
    gemm(1, xq_bf, 0, hwT_bf, 0, hw_t_b, t_q,    0, (int)BJ, 400, KD, 400, 1);
    gemm(2, xq_bf, 0, hwG_bf, 0, hw_g_b, gate_q, 0, (int)BJ, 400, KD, 400, 1);
    hw_combine<<<nb(BT * 400, THR), THR, 0, stream>>>(x_c, t_c, gate_c, xh_c, BT * 400);
    hw_combine<<<nb(BJ * 400, THR), THR, 0, stream>>>(x_q, t_q, gate_q, xh_q, BJ * 400);

    // ---- contextual BiGRU (context then query)
    cvt_pad_bf16<<<nb(BT * KD, THR), THR, 0, stream>>>(xh_c, xc_bf, BT, 400, KD);
    cvt_pad_bf16<<<nb(BJ * KD, THR), THR, 0, stream>>>(xh_q, xq_bf, BJ, 400, KD);
    gemm(0, xc_bf, 0, WihCtxF, 0, ctxF_bih, xp_f, 0, (int)BT, 1200, KD, 1200, 1);
    gemm(0, xc_bf, 0, WihCtxB, 0, ctxB_bih, xp_b, 0, (int)BT, 1200, KD, 1200, 1);
    gru(xp_f, xp_b, WhhCtxF, WhhCtxB, ctxF_bhh, ctxB_bhh, c_ctx, T);
    gemm(0, xq_bf, 0, WihCtxF, 0, ctxF_bih, xp_f, 0, (int)BJ, 1200, KD, 1200, 1);
    gemm(0, xq_bf, 0, WihCtxB, 0, ctxB_bih, xp_b, 0, (int)BJ, 1200, KD, 1200, 1);
    gru(xp_f, xp_b, WhhCtxF, WhhCtxB, ctxF_bhh, ctxB_bhh, q_ctx, J);

    // ---- attention flow
    rowdot<<<nb(BT, THR), THR, 0, stream>>>(c_ctx, w_s,        cwh, BT, 800);
    rowdot<<<nb(BJ, THR), THR, 0, stream>>>(q_ctx, w_s + 800,  qwu, BJ, 800);
    scale_cvt_bf16<<<nb(BT * 800, THR), THR, 0, stream>>>(c_ctx, w_s + 1600, cw_bf, BT, 800);
    cvt_pad_bf16<<<nb(BJ * 800, THR), THR, 0, stream>>>(q_ctx, q_bf, BJ, 800, 800);
    gemm(0, cw_bf, (long long)T * 800, q_bf, (long long)J * 800, nullptr,
         sim, (long long)T * J, T, J, 800, J, B);
    sim_add<<<nb((long long)B * T * J, THR), THR, 0, stream>>>(sim, cwh, qwu, B, T, J);
    rowmax_J<<<nb(BT, THR), THR, 0, stream>>>(sim, smax, BT, J);
    softmax_T_bj<<<nb((long long)B * J, THR), THR, 0, stream>>>(sim, B, T, J);  // sim -> att_a
    cvt_pad_bf16<<<nb(BT * J, THR), THR, 0, stream>>>(sim, atta_bf, BT, J, J);
    transpose_bf16_batched<<<nb((long long)B * J * 800, THR), THR, 0, stream>>>(q_ctx, qT_bf, B, J, 800);
    gemm(0, atta_bf, (long long)T * J, qT_bf, 800LL * J, nullptr,
         u_t, (long long)T * 800, T, 800, J, 800, B);
    softmax_T_b<<<1, 32, 0, stream>>>(smax, attb, B, T);
    weighted_sum_T<<<nb((long long)B * 800, THR), THR, 0, stream>>>(attb, c_ctx, h_t, B, T, 800);
    build_g<<<nb(BT * 800, THR), THR, 0, stream>>>(c_ctx, u_t, h_t, gbuf, B, T, 800);

    // ---- modeling layer 1 (input 3200)
    cvt_pad_bf16<<<nb(BT * 3200, THR), THR, 0, stream>>>(gbuf, g_bf, BT, 3200, 3200);
    gemm(0, g_bf, 0, WihM1F, 0, m1F_bih, xp_f, 0, (int)BT, 1200, 3200, 1200, 1);
    gemm(0, g_bf, 0, WihM1B, 0, m1B_bih, xp_b, 0, (int)BT, 1200, 3200, 1200, 1);
    gru(xp_f, xp_b, WhhM1F, WhhM1B, m1F_bhh, m1B_bhh, m1, T);

    // ---- modeling layer 2 (input 800)
    cvt_pad_bf16<<<nb(BT * 800, THR), THR, 0, stream>>>(m1, m1_bf, BT, 800, 800);
    gemm(0, m1_bf, 0, WihM2F, 0, m2F_bih, xp_f, 0, (int)BT, 1200, 800, 1200, 1);
    gemm(0, m1_bf, 0, WihM2B, 0, m2B_bih, xp_b, 0, (int)BT, 1200, 800, 1200, 1);
    gru(xp_f, xp_b, WhhM2F, WhhM2B, m2F_bhh, m2B_bhh, mbuf, T);

    // ---- p_start
    score_dot<<<nb(BT, 128), 128, 0, stream>>>(gbuf, mbuf, w_p_start, s1, BT);
    softmax_T_b<<<1, 32, 0, stream>>>(s1, (float*)d_out, B, T);

    // ---- output layer BiGRU + p_end
    cvt_pad_bf16<<<nb(BT * 800, THR), THR, 0, stream>>>(mbuf, m_bf, BT, 800, 800);
    gemm(0, m_bf, 0, WihOF, 0, oF_bih, xp_f, 0, (int)BT, 1200, 800, 1200, 1);
    gemm(0, m_bf, 0, WihOB, 0, oB_bih, xp_b, 0, (int)BT, 1200, 800, 1200, 1);
    gru(xp_f, xp_b, WhhOF, WhhOB, oF_bhh, oB_bhh, m2, T);
    score_dot<<<nb(BT, 128), 128, 0, stream>>>(gbuf, m2, w_p_end, s2, BT);
    softmax_T_b<<<1, 32, 0, stream>>>(s2, (float*)d_out + BT, B, T);
}
